// W8A16LinearLayer_7112465842777
// MI455X (gfx1250) — matrix operand order repaired and checked by reading, not compiled, not run on hardware
//
#include <hip/hip_runtime.h>

typedef __attribute__((ext_vector_type(16))) _Float16 v16h;
typedef __attribute__((ext_vector_type(8)))  _Float16 v8h;
typedef __attribute__((ext_vector_type(2)))  _Float16 v2h;
typedef __attribute__((ext_vector_type(8)))  float  v8f;
typedef __attribute__((ext_vector_type(4)))  float  v4f;
typedef __attribute__((ext_vector_type(16))) signed char v16c;

#define IN_F   4096
#define OUT_F  11008
#define MROWS  8192            // B*S = 4*2048

#define TILE_M 256
#define TILE_N 128
#define TILE_K 64
#define LDSK   72              // padded K stride: 144B rows, conflict-free b128
#define NKT    (IN_F / TILE_K) // 64 K-steps

// Exact int8 -> f16 for 4 packed bytes: f16(b) = as_f16(0x6400 | (b^0x80)) - 1152
// (1024 + u is integer-exact in f16 for u in [0,255])
__device__ __forceinline__ void cvt4_i8_f16(unsigned w, v2h* out2) {
    const unsigned u = w ^ 0x80808080u;
    const v2h off = {(_Float16)1152.0f, (_Float16)1152.0f};
    union { unsigned u; v2h h; } c0, c1;
    c0.u = __builtin_amdgcn_perm(0x64646464u, u, 0x04010400u); // {64,b1,64,b0}
    c1.u = __builtin_amdgcn_perm(0x64646464u, u, 0x04030402u); // {64,b3,64,b2}
    out2[0] = c0.h - off;
    out2[1] = c1.h - off;
}

__global__ __launch_bounds__(256)
void w8a16_gemm_f16_wmma(const float* __restrict__ X,
                         const signed char* __restrict__ W,
                         const float* __restrict__ scales,
                         const float* __restrict__ bias,
                         float* __restrict__ out)
{
    __shared__ _Float16 As[2][TILE_M * LDSK];   // [buf][row m][k]
    __shared__ _Float16 Bs[2][TILE_N * LDSK];   // [buf][col n][k]

    const int tid  = threadIdx.x;
    const int lane = tid & 31;
    const int wv   = tid >> 5;       // 0..7
    const int wm   = wv >> 1;        // wave row 0..3 (64 M-rows each)
    const int wn   = wv & 1;         // wave col 0..1 (64 N-cols each)
    const int kh   = lane >> 4;      // half-wave select
    const int lr   = lane & 15;

    const int bm = blockIdx.x * TILE_M;
    const int bn = blockIdx.y * TILE_N;

    // staging: A = 256x64 fp32 (8 thr/row -> 32B each, 8 passes of 32 rows)
    //          B = 128x64 int8 (2 thr/row -> 32B each, single pass)
    const int a_row = tid >> 3;
    const int a_kc  = (tid & 7) * 8;
    const int b_row = tid >> 1;
    const int b_kc  = (tid & 1) * 32;

    const float*       Xp = X + (size_t)(bm + a_row) * IN_F + a_kc;
    const signed char* Wp = W + (size_t)(bn + b_row) * IN_F + b_kc;

    v4f ax[16];
    union { v16c c; unsigned d[4]; } bx[2];

    auto load_tile = [&](int k0) {
        #pragma unroll
        for (int p = 0; p < 8; ++p) {
            ax[2 * p]     = *(const v4f*)(Xp + (size_t)p * 32 * IN_F + k0);
            ax[2 * p + 1] = *(const v4f*)(Xp + (size_t)p * 32 * IN_F + k0 + 4);
        }
        bx[0].c = *(const v16c*)(Wp + k0);
        bx[1].c = *(const v16c*)(Wp + k0 + 16);
    };

    auto store_tile = [&](int buf) {
        #pragma unroll
        for (int p = 0; p < 8; ++p) {
            v8h t;
            #pragma unroll
            for (int e = 0; e < 4; ++e) {
                t[e]     = (_Float16)ax[2 * p][e];
                t[e + 4] = (_Float16)ax[2 * p + 1][e];
            }
            *(v8h*)&As[buf][(p * 32 + a_row) * LDSK + a_kc] = t;
        }
        #pragma unroll
        for (int c = 0; c < 2; ++c) {
            union { v8h v; v2h p2[4]; } wlo, whi;
            cvt4_i8_f16(bx[c].d[0], &wlo.p2[0]);
            cvt4_i8_f16(bx[c].d[1], &wlo.p2[2]);
            cvt4_i8_f16(bx[c].d[2], &whi.p2[0]);
            cvt4_i8_f16(bx[c].d[3], &whi.p2[2]);
            _Float16* dst = &Bs[buf][b_row * LDSK + b_kc + c * 16];
            *(v8h*)dst       = wlo.v;
            *(v8h*)(dst + 8) = whi.v;
        }
    };

    v8f acc[4][4];
    {
        v8f z = {};
        #pragma unroll
        for (int i = 0; i < 4; ++i)
            #pragma unroll
            for (int j = 0; j < 4; ++j) acc[i][j] = z;
    }

    auto compute = [&](int buf) {
        #pragma unroll
        for (int ks = 0; ks < 2; ++ks) {   // two 16x16x32 K-subtiles
            // A 16x32 f16: lane(0-15)=row M, K chunks [kh*8,+8) and [16+kh*8,+8)
            v16h af[4];
            const _Float16* Ab = &As[buf][(wm * 64 + lr) * LDSK + ks * 32 + kh * 8];
            #pragma unroll
            for (int i = 0; i < 4; ++i) {
                union { v16h v; v8h h[2]; } ua;
                ua.h[0] = *(const v8h*)(Ab + i * 16 * LDSK);
                ua.h[1] = *(const v8h*)(Ab + i * 16 * LDSK + 16);
                af[i] = ua.v;
            }
            // B 32x16 f16: lane = column N, K chunks [kh*8,+8) and [16+kh*8,+8)
            v16h bf[4];
            const _Float16* Bb = &Bs[buf][(wn * 64 + lr) * LDSK + ks * 32 + kh * 8];
            #pragma unroll
            for (int j = 0; j < 4; ++j) {
                union { v16h v; v8h h[2]; } ub;
                ub.h[0] = *(const v8h*)(Bb + j * 16 * LDSK);
                ub.h[1] = *(const v8h*)(Bb + j * 16 * LDSK + 16);
                bf[j] = ub.v;
            }
            #pragma unroll
            for (int i = 0; i < 4; ++i)
                #pragma unroll
                for (int j = 0; j < 4; ++j)
                    acc[i][j] = __builtin_amdgcn_wmma_f32_16x16x32_f16(
                        false, af[i], false, bf[j],
                        (short)0, acc[i][j], false, false);
        }
    };

    // ---- software pipeline: double-buffered LDS, one barrier per K-step ----
    load_tile(0);
    store_tile(0);
    __syncthreads();

    for (int kt = 0; kt < NKT; ++kt) {
        const int cur = kt & 1;
        if (kt + 1 < NKT) load_tile((kt + 1) * TILE_K);   // overlap with compute
        if (kt + 2 < NKT) {                                // L2 prefetch 2 ahead
            __builtin_prefetch(Xp + (kt + 2) * TILE_K, 0, 1);
            __builtin_prefetch(Wp + (kt + 2) * TILE_K, 0, 1);
        }
        compute(cur);
        if (kt + 1 < NKT) store_tile(cur ^ 1);             // fill other buffer
        __syncthreads();
    }

    // ---- epilogue: per-channel scale + bias, coalesced row stores ----
    #pragma unroll
    for (int j = 0; j < 4; ++j) {
        const int n = bn + wn * 64 + j * 16 + lr;
        const float s  = scales[n];
        const float bz = bias[n];
        #pragma unroll
        for (int i = 0; i < 4; ++i) {
            const int m = bm + wm * 64 + i * 16 + kh * 8;
            #pragma unroll
            for (int r = 0; r < 8; ++r) {
                out[(size_t)(m + r) * OUT_F + n] = acc[i][j][r] * s + bz;
            }
        }
    }
}

extern "C" void kernel_launch(void* const* d_in, const int* in_sizes, int n_in,
                              void* d_out, int out_size, void* d_ws, size_t ws_size,
                              hipStream_t stream) {
    const float*       X  = (const float*)d_in[0];
    const signed char* W  = (const signed char*)d_in[1];
    const float*       Sc = (const float*)d_in[2];
    const float*       Bz = (const float*)d_in[3];
    float*             O  = (float*)d_out;

    dim3 grid(MROWS / TILE_M, OUT_F / TILE_N);   // 32 x 86
    w8a16_gemm_f16_wmma<<<grid, 256, 0, stream>>>(X, W, Sc, Bz, O);
}
